// InternLM2Attention_32607391711250
// MI455X (gfx1250) — matrix operand order repaired and checked by reading, not compiled, not run on hardware
//
#include <hip/hip_runtime.h>
#include <hip/hip_bf16.h>
#include <stdint.h>

// ---------------------------------------------------------------------------
// InternLM2 attention block (PLoRA qkv -> RoPE -> GQA flash attn -> PLoRA wo)
// MI455X / gfx1250: wave32, V_WMMA_F32_16X16X32_BF16 for all matmuls.
// GEMM waves compute 16x128 tiles (8 WMMA N-tiles per A fragment) to maximize
// WMMA density per byte loaded; weights stream once (~90 MB bf16 vs 23.3 TB/s),
// activations live in the 192 MB L2.
// ---------------------------------------------------------------------------

typedef __bf16 v16bf __attribute__((ext_vector_type(16)));
typedef __bf16 v8bf  __attribute__((ext_vector_type(8)));
typedef float  v8f   __attribute__((ext_vector_type(8)));

static __device__ inline v8f vzero8() {
  v8f z = {0.f, 0.f, 0.f, 0.f, 0.f, 0.f, 0.f, 0.f};
  return z;
}

static __device__ inline v8f wmma_bf16(v16bf a, v16bf b, v8f c) {
  // (neg_a, A, neg_b, B, c_mod, C, reuse_a, reuse_b)
  return __builtin_amdgcn_wmma_f32_16x16x32_bf16(false, a, false, b, (short)0, c,
                                                 false, false);
}

static __device__ inline v16bf pack16(v8bf lo, v8bf hi) {
  v16bf r;
#pragma unroll
  for (int i = 0; i < 8; ++i) {
    r[i] = lo[i];
    r[i + 8] = hi[i];
  }
  return r;
}

// ---------------------------------------------------------------------------
// fp32 -> bf16 conversion (grid-stride)
// ---------------------------------------------------------------------------
__global__ void k_cvt_bf16(const float* __restrict__ src,
                           __bf16* __restrict__ dst, size_t n) {
  size_t i = (size_t)blockIdx.x * blockDim.x + threadIdx.x;
  size_t stride = (size_t)gridDim.x * blockDim.x;
  for (; i < n; i += stride) dst[i] = (__bf16)src[i];
}

// ---------------------------------------------------------------------------
// C[M,N] = A[M,K] . B[N,K]^T   (A,B bf16 row-major, C fp32)
// 128 threads = 4 waves; wave w -> 16 rows, 128 cols (8 WMMA N-tiles, A reuse).
// A fragment (16x32 bf16): lane l holds row (l&15), K chunk (l>>4)*16.
// B fragment (32x16 bf16): lane l holds col (l&15) = row of B, K chunk (l>>4)*16.
// ---------------------------------------------------------------------------
__global__ __launch_bounds__(128) void k_gemm_bf16(
    const __bf16* __restrict__ A, const __bf16* __restrict__ B,
    float* __restrict__ C, int M, int N, int K) {
  const int wave = threadIdx.x >> 5;
  const int lane = threadIdx.x & 31;
  const int lan16 = lane & 15;
  const int half = lane >> 4;
  const int m0 = (blockIdx.y * 4 + wave) * 16;
  const int n0 = blockIdx.x * 128;

  const __bf16* Arow = A + (size_t)(m0 + lan16) * K + half * 16;
  v8f acc[8];
#pragma unroll
  for (int t = 0; t < 8; ++t) acc[t] = vzero8();

  for (int k0 = 0; k0 < K; k0 += 32) {
    if (k0 + 512 < K) __builtin_prefetch(Arow + k0 + 512, 0, 1);
    v16bf a = *(const v16bf*)(Arow + k0);
#pragma unroll
    for (int t = 0; t < 8; ++t) {
      const __bf16* pb =
          B + (size_t)(n0 + t * 16 + lan16) * K + k0 + half * 16;
      v16bf b = *(const v16bf*)pb;
      acc[t] = wmma_bf16(a, b, acc[t]);
    }
  }
#pragma unroll
  for (int t = 0; t < 8; ++t)
#pragma unroll
    for (int r = 0; r < 8; ++r)
      C[(size_t)(m0 + half * 8 + r) * N + n0 + t * 16 + lan16] = acc[t][r];
}

// Same GEMM, but accumulates into C only on rows where rowmask != 0 (PLoRA).
__global__ __launch_bounds__(128) void k_gemm_bf16_acc_mask(
    const __bf16* __restrict__ A, const __bf16* __restrict__ B,
    float* __restrict__ C, const unsigned char* __restrict__ rowmask, int M,
    int N, int K) {
  const int wave = threadIdx.x >> 5;
  const int lane = threadIdx.x & 31;
  const int lan16 = lane & 15;
  const int half = lane >> 4;
  const int m0 = (blockIdx.y * 4 + wave) * 16;
  const int n0 = blockIdx.x * 128;

  const __bf16* Arow = A + (size_t)(m0 + lan16) * K + half * 16;
  v8f acc[8];
#pragma unroll
  for (int t = 0; t < 8; ++t) acc[t] = vzero8();

  for (int k0 = 0; k0 < K; k0 += 32) {
    v16bf a = *(const v16bf*)(Arow + k0);
#pragma unroll
    for (int t = 0; t < 8; ++t) {
      const __bf16* pb =
          B + (size_t)(n0 + t * 16 + lan16) * K + k0 + half * 16;
      v16bf b = *(const v16bf*)pb;
      acc[t] = wmma_bf16(a, b, acc[t]);
    }
  }
#pragma unroll
  for (int r = 0; r < 8; ++r) {
    const int row = m0 + half * 8 + r;
    if (rowmask[row]) {
#pragma unroll
      for (int t = 0; t < 8; ++t)
        C[(size_t)row * N + n0 + t * 16 + lan16] += acc[t][r];
    }
  }
}

// ---------------------------------------------------------------------------
// RoPE + head split. Y1 viewed as [S, 8(kv), 6(slot), 128].
// slot 0..3 -> Q head kv*4+slot (RoPE), slot 4 -> K (RoPE), slot 5 -> Vt.
// Q: [32][S][128] bf16, K: [8][S][128] bf16, Vt: [8][128][S] bf16 (transposed
// so P.V WMMA B-fragments are contiguous loads).
// ---------------------------------------------------------------------------
__global__ void k_rope_split(const float* __restrict__ Y,
                             const long long* __restrict__ pos_ids,
                             __bf16* __restrict__ Qb, __bf16* __restrict__ Kb,
                             __bf16* __restrict__ Vt) {
  const size_t total = (size_t)2048 * 8 * 6 * 128;
  size_t tid = (size_t)blockIdx.x * blockDim.x + threadIdx.x;
  if (tid >= total) return;
  const int d = (int)(tid & 127);
  size_t t = tid >> 7;
  const int slot = (int)(t % 6);
  t /= 6;
  const int kv = (int)(t % 8);
  const int s = (int)(t / 8);
  const size_t base = (((size_t)s * 8 + kv) * 6 + slot) * 128;
  const float v = Y[base + d];
  if (slot == 5) {
    Vt[((size_t)kv * 128 + d) * 2048 + s] = (__bf16)v;
    return;
  }
  const float partner = Y[base + ((d < 64) ? d + 64 : d - 64)];
  const float rh = (d < 64) ? -partner : partner;
  const int i = d & 63;
  const float pos = (float)pos_ids[s];
  const float ang = pos * __powf(10000.0f, -(float)i * (1.0f / 64.0f));
  const float c = __cosf(ang);
  const float sn = __sinf(ang);
  const float outv = v * c + rh * sn;
  if (slot == 4) {
    Kb[((size_t)kv * 2048 + s) * 128 + d] = (__bf16)outv;
  } else {
    const int h = kv * 4 + slot;
    Qb[((size_t)h * 2048 + s) * 128 + d] = (__bf16)outv;
  }
}

// ---------------------------------------------------------------------------
// Causal GQA flash attention. One wave per (head, 16-query tile).
// kv step 32: two 16x16 score WMMAs (K over d in 4 chunks of 32), online
// softmax in fp32 (row stats reduced across the 16-lane half-wave holding each
// row), P re-shaped C->A layout via 1KB LDS, then 8 P.V WMMAs (d = 8 x 16).
// kv0 is a multiple of 32 and keys <= 2047 -> no OOB loads; causal handled by
// -1e30 masking (exp -> 0), matching the reference's -1e9 additive mask.
// ---------------------------------------------------------------------------
__global__ __launch_bounds__(32) void k_attn(const __bf16* __restrict__ Q,
                                             const __bf16* __restrict__ Kc,
                                             const __bf16* __restrict__ Vt,
                                             __bf16* __restrict__ O) {
  const int S = 2048, D = 128;
  const int qt = blockIdx.x;
  const int h = blockIdx.y;
  const int q0 = qt * 16;
  const int kvh = h >> 2;  // repeat_interleave(g=4)
  const int lane = threadIdx.x;
  const int lan16 = lane & 15;
  const int half = lane >> 4;

  const __bf16* Qh = Q + (size_t)h * S * D;
  const __bf16* Kh = Kc + (size_t)kvh * S * D;
  const __bf16* Vh = Vt + (size_t)kvh * D * S;

  v16bf aQ[4];
#pragma unroll
  for (int j = 0; j < 4; ++j) {
    const __bf16* p = Qh + (size_t)(q0 + lan16) * D + j * 32 + half * 16;
    aQ[j] = *(const v16bf*)p;
  }

  v8f acc[8];
#pragma unroll
  for (int t = 0; t < 8; ++t) acc[t] = vzero8();
  float mi[8], li[8];
#pragma unroll
  for (int r = 0; r < 8; ++r) {
    mi[r] = -3.0e38f;
    li[r] = 0.0f;
  }

  __shared__ __align__(64) __bf16 ldsP[16 * 32];
  const float scale = 0.08838834764831845f;  // 1/sqrt(128)

  for (int kv0 = 0; kv0 < q0 + 16; kv0 += 32) {
    v8f sc[2];
#pragma unroll
    for (int hh = 0; hh < 2; ++hh) {
      const int kb = kv0 + hh * 16;
      v8f c = vzero8();
#pragma unroll
      for (int j = 0; j < 4; ++j) {
        const __bf16* pk = Kh + (size_t)(kb + lan16) * D + j * 32 + half * 16;
        v16bf b = *(const v16bf*)pk;
        c = wmma_bf16(aQ[j], b, c);
      }
      const int key = kb + lan16;
#pragma unroll
      for (int r = 0; r < 8; ++r) {
        const int row = q0 + half * 8 + r;
        sc[hh][r] = (key <= row) ? c[r] * scale : -1.0e30f;
      }
    }
#pragma unroll
    for (int r = 0; r < 8; ++r) {
      float s0 = sc[0][r], s1 = sc[1][r];
      float mx = fmaxf(s0, s1);
#pragma unroll
      for (int off = 1; off < 16; off <<= 1)
        mx = fmaxf(mx, __shfl_xor(mx, off, 16));
      const float mnew = fmaxf(mi[r], mx);
      const float alpha = __expf(mi[r] - mnew);
      const float p0 = __expf(s0 - mnew);
      const float p1 = __expf(s1 - mnew);
      float rs = p0 + p1;
#pragma unroll
      for (int off = 1; off < 16; off <<= 1) rs += __shfl_xor(rs, off, 16);
      li[r] = li[r] * alpha + rs;
      mi[r] = mnew;
#pragma unroll
      for (int t = 0; t < 8; ++t) acc[t][r] *= alpha;
      ldsP[(half * 8 + r) * 32 + lan16] = (__bf16)p0;
      ldsP[(half * 8 + r) * 32 + 16 + lan16] = (__bf16)p1;
    }
    __syncthreads();
    const __bf16* lp = ldsP + lan16 * 32 + half * 16;
    v16bf aP = *(const v16bf*)lp;
    __syncthreads();
#pragma unroll
    for (int t = 0; t < 8; ++t) {
      const __bf16* pv = Vh + (size_t)(t * 16 + lan16) * S + kv0 + half * 16;
      v16bf b = *(const v16bf*)pv;
      acc[t] = wmma_bf16(aP, b, acc[t]);
    }
  }

#pragma unroll
  for (int r = 0; r < 8; ++r) {
    const int row = q0 + half * 8 + r;
    const float inv = 1.0f / li[r];
#pragma unroll
    for (int t = 0; t < 8; ++t)
      O[(size_t)row * 4096 + h * 128 + t * 16 + lan16] =
          (__bf16)(acc[t][r] * inv);
  }
}

// ---------------------------------------------------------------------------
// Launcher
// ---------------------------------------------------------------------------
extern "C" void kernel_launch(void* const* d_in, const int* in_sizes, int n_in,
                              void* d_out, int out_size, void* d_ws,
                              size_t ws_size, hipStream_t stream) {
  (void)in_sizes;
  (void)n_in;
  (void)out_size;
  (void)ws_size;

  const float* X = (const float*)d_in[0];
  // d_in[1] = attention_mask (causal additive) -> handled analytically
  const float* wqkv_w = (const float*)d_in[2];
  const float* wqkv_a = (const float*)d_in[3];
  const float* wqkv_b = (const float*)d_in[4];
  const float* wo_w = (const float*)d_in[5];
  const float* wo_a = (const float*)d_in[6];
  const float* wo_b = (const float*)d_in[7];
  const unsigned char* im_mask = (const unsigned char*)d_in[8];
  const long long* pos = (const long long*)d_in[9];
  float* out = (float*)d_out;

  const int S = 2048, H = 4096, QKV = 6144, R = 256;

  char* ws = (char*)d_ws;
  auto carve = [&](size_t bytes) -> char* {
    char* p = ws;
    ws += (bytes + 255) & ~(size_t)255;
    return p;
  };

  __bf16* Xb = (__bf16*)carve((size_t)S * H * 2);
  __bf16* W1b = (__bf16*)carve((size_t)QKV * H * 2);
  __bf16* A1b = (__bf16*)carve((size_t)R * H * 2);
  __bf16* B1b = (__bf16*)carve((size_t)QKV * R * 2);
  __bf16* W2b = (__bf16*)carve((size_t)H * H * 2);
  __bf16* A2b = (__bf16*)carve((size_t)R * H * 2);
  __bf16* B2b = (__bf16*)carve((size_t)H * R * 2);
  float* Y1 = (float*)carve((size_t)S * QKV * 4);
  float* T = (float*)carve((size_t)S * R * 4);
  __bf16* Tb = (__bf16*)carve((size_t)S * R * 2);
  __bf16* Qb = (__bf16*)carve((size_t)32 * S * 128 * 2);
  __bf16* Kb = (__bf16*)carve((size_t)8 * S * 128 * 2);
  __bf16* Vt = (__bf16*)carve((size_t)8 * 128 * S * 2 + 256);
  __bf16* Cb = (__bf16*)carve((size_t)S * H * 2);

  auto cvt = [&](const float* s, __bf16* d, size_t n) {
    hipLaunchKernelGGL(k_cvt_bf16, dim3((unsigned)((n + 255) / 256)), dim3(256),
                       0, stream, s, d, n);
  };

  // 1. bf16 conversions (weights stream once; X fits in L2 as bf16)
  cvt(X, Xb, (size_t)S * H);
  cvt(wqkv_w, W1b, (size_t)QKV * H);
  cvt(wqkv_a, A1b, (size_t)R * H);
  cvt(wqkv_b, B1b, (size_t)QKV * R);
  cvt(wo_w, W2b, (size_t)H * H);
  cvt(wo_a, A2b, (size_t)R * H);
  cvt(wo_b, B2b, (size_t)H * R);

  // 2. PLoRA qkv: Y1 = X.Wqkv^T + im_mask * (X.A^T).B^T
  hipLaunchKernelGGL(k_gemm_bf16, dim3(R / 128, S / 64), dim3(128), 0, stream,
                     Xb, A1b, T, S, R, H);
  cvt(T, Tb, (size_t)S * R);
  hipLaunchKernelGGL(k_gemm_bf16, dim3(QKV / 128, S / 64), dim3(128), 0, stream,
                     Xb, W1b, Y1, S, QKV, H);
  hipLaunchKernelGGL(k_gemm_bf16_acc_mask, dim3(QKV / 128, S / 64), dim3(128),
                     0, stream, Tb, B1b, Y1, im_mask, S, QKV, R);

  // 3. RoPE + split into Q/K/Vt (bf16)
  {
    size_t n = (size_t)S * 8 * 6 * 128;
    hipLaunchKernelGGL(k_rope_split, dim3((unsigned)((n + 255) / 256)),
                       dim3(256), 0, stream, Y1, pos, Qb, Kb, Vt);
  }

  // 4. Causal GQA flash attention -> context Cb [S, 4096] bf16
  hipLaunchKernelGGL(k_attn, dim3(S / 16, 32), dim3(32), 0, stream, Qb, Kb, Vt,
                     Cb);

  // 5. PLoRA output projection straight into d_out (fp32)
  hipLaunchKernelGGL(k_gemm_bf16, dim3(R / 128, S / 64), dim3(128), 0, stream,
                     Cb, A2b, T, S, R, H);
  cvt(T, Tb, (size_t)S * R);
  hipLaunchKernelGGL(k_gemm_bf16, dim3(H / 128, S / 64), dim3(128), 0, stream,
                     Cb, W2b, out, S, H, H);
  hipLaunchKernelGGL(k_gemm_bf16_acc_mask, dim3(H / 128, S / 64), dim3(128), 0,
                     stream, Tb, B2b, out, im_mask, S, H, R);
}
